// GNN_3006477107608
// MI455X (gfx1250) — matrix operand order repaired and checked by reading, not compiled, not run on hardware
//
#include <hip/hip_runtime.h>
#include <math.h>

#define DIM 128

typedef _Float16 v8h  __attribute__((ext_vector_type(8)));
typedef _Float16 v16h __attribute__((ext_vector_type(16)));
typedef float    v8f  __attribute__((ext_vector_type(8)));
typedef unsigned tdm_u4 __attribute__((ext_vector_type(4)));
typedef int      tdm_i4 __attribute__((ext_vector_type(4)));
typedef int      tdm_i8 __attribute__((ext_vector_type(8)));

// ---------- device helpers ----------
__device__ __forceinline__ float fatomic_add(float* p, float v) {
  return __hip_atomic_fetch_add(p, v, __ATOMIC_RELAXED, __HIP_MEMORY_SCOPE_AGENT);
}
__device__ __forceinline__ unsigned uatomic_max(unsigned* p, unsigned v) {
  return __hip_atomic_fetch_max(p, v, __ATOMIC_RELAXED, __HIP_MEMORY_SCOPE_AGENT);
}
// order-preserving float<->uint map (works for negatives); 0u acts as -inf sentinel
__device__ __forceinline__ unsigned ordf(float f) {
  unsigned u = __float_as_uint(f);
  return (u & 0x80000000u) ? ~u : (u | 0x80000000u);
}
__device__ __forceinline__ float unordf(unsigned o) {
  unsigned u = (o & 0x80000000u) ? (o & 0x7fffffffu) : ~o;
  return __uint_as_float(u);
}

// ---------- utility ----------
__global__ void fill_u32(unsigned* p, unsigned v, int n) {
  int i = blockIdx.x * blockDim.x + threadIdx.x;
  if (i < n) p[i] = v;
}

// deg[n] = #edges with dst == n (as float); shared by both layers
__global__ void deg_kernel(const int* dst, float* deg, int E) {
  int e = blockIdx.x * blockDim.x + threadIdx.x;
  if (e < E) fatomic_add(deg + dst[e], 1.0f);
}

// kred[d] = sum_n deg[n] * X[n,d]   (column reduction, 64 rows per block)
__global__ void kred_kernel(const float* __restrict__ X, const float* __restrict__ deg,
                            float* kred, int n) {
  int col = threadIdx.x;           // blockDim.x == 128
  int r0 = blockIdx.x * 64;
  int r1 = min(r0 + 64, n);
  float acc = 0.f;
  for (int r = r0; r < r1; ++r) acc += deg[r] * X[(size_t)r * DIM + col];
  fatomic_add(kred + col, acc);
}

// Tiny per-layer vector algebra (single block, 128 threads):
//   ksum = kred @ kw^T + E*kb ;  t = W @ ksum ;  u = qw^T @ t * s ;  c = (qb . t) * s
__global__ void vecmath_kernel(const float* __restrict__ kred,
                               const float* __restrict__ kw, const float* __restrict__ kb,
                               const float* __restrict__ W,
                               const float* __restrict__ qw, const float* __restrict__ qb,
                               float degsum, float scale, float* u, float* c) {
  __shared__ float kr[DIM], ks[DIM], t[DIM];
  int d = threadIdx.x;
  kr[d] = kred[d];
  __syncthreads();
  float a = 0.f;
  for (int j = 0; j < DIM; ++j) a += kr[j] * kw[d * DIM + j];
  ks[d] = a + degsum * kb[d];
  __syncthreads();
  a = 0.f;
  for (int j = 0; j < DIM; ++j) a += W[d * DIM + j] * ks[j];
  t[d] = a;
  __syncthreads();
  a = 0.f;
  for (int i = 0; i < DIM; ++i) a += qw[i * DIM + d] * t[i];
  u[d] = a * scale;
  if (d == 0) {
    float cc = 0.f;
    for (int i = 0; i < DIM; ++i) cc += qb[i] * t[i];
    *c = cc * scale;
  }
}

// score[n] = X[n,:] . u + c    (one wave32 per row; float4/lane, coalesced 512B)
__global__ void score_kernel(const float* __restrict__ X, const float* __restrict__ u,
                             const float* __restrict__ c, float* score, int n) {
  int w = (blockIdx.x * blockDim.x + threadIdx.x) >> 5;
  int lane = threadIdx.x & 31;
  if (w >= n) return;
  float4 xv = ((const float4*)(X + (size_t)w * DIM))[lane];
  float4 uv = ((const float4*)u)[lane];
  float p = xv.x * uv.x + xv.y * uv.y + xv.z * uv.z + xv.w * uv.w;
  for (int o = 16; o > 0; o >>= 1) p += __shfl_down(p, o, 32);
  if (lane == 0) score[w] = p + *c;
}

// f32 -> f16 pack of activations
__global__ void pack_x_kernel(const float* __restrict__ X, _Float16* Xh, int n) {
  int i = blockIdx.x * blockDim.x + threadIdx.x;
  if (i < n) Xh[i] = (_Float16)X[i];
}

// Bh[k*128+n] = vw[n*128+k]  (B = vw^T, f16, row-major over K)
__global__ void pack_w_kernel(const float* __restrict__ vw, _Float16* Bh) {
  int i = blockIdx.x * blockDim.x + threadIdx.x;   // 16384
  if (i >= DIM * DIM) return;
  int k = i >> 7, n = i & 127;
  Bh[i] = (_Float16)vw[n * DIM + k];
}

// V = Xh @ Bh + vb  via v_wmma_f32_16x16x32_f16, with the full 32KB B matrix
// staged into LDS once per block by the Tensor Data Mover (TENSORcnt-tracked).
// Each wave computes TWO 16-row M-tiles across the full 128-wide N:
// 2 independent WMMA chains, A from global (b128), B from LDS (strided 16-bit reads, one per element).
__global__ void __launch_bounds__(256) vproj_wmma(const _Float16* __restrict__ Xh,
                                                  const _Float16* __restrict__ Bh,
                                                  const float* __restrict__ vb,
                                                  float* __restrict__ V, int n_mtiles) {
  __shared__ __attribute__((aligned(32))) _Float16 Bs[DIM * DIM];  // 32 KB of 320 KB LDS

#if __has_builtin(__builtin_amdgcn_tensor_load_to_lds)
  if (threadIdx.x < 32) {  // one wave issues the TDM descriptor (TDM ignores EXEC)
    unsigned lds_off = (unsigned)(size_t)(&Bs[0]);      // generic addr low 32 bits == LDS offset
    unsigned long long ga = (unsigned long long)(size_t)Bh;
    tdm_u4 g0;
    g0.x = 1u;                                          // count=1 (valid), gather_mode=0
    g0.y = lds_off;                                     // lds_addr
    g0.z = (unsigned)ga;                                // global_addr[31:0]
    g0.w = (unsigned)((ga >> 32) & 0x1FFFFFFu) | (2u << 30);  // global_addr[56:32] | type=2
    tdm_i8 g1;
    g1[0] = 0x00010000;            // workgroup_mask=0 (not in cluster), data_size=1 (2 bytes)
    g1[1] = (int)(128u << 16);     // tensor_dim0[15:0]=128 in bits[31:16]
    g1[2] = (int)(128u << 16);     // tensor_dim0 hi=0 ; tensor_dim1[15:0]=128 in bits[31:16]
    g1[3] = (int)(128u << 16);     // tensor_dim1 hi=0 ; tile_dim0=128 in bits[31:16]
    g1[4] = 128;                   // tile_dim1=128 ; tile_dim2=0
    g1[5] = 128;                   // tensor_dim0_stride[31:0]=128
    g1[6] = 0;                     // stride hi / dim1_stride lo
    g1[7] = 0;
    tdm_i4 gz = {0, 0, 0, 0};      // 2-D tensor: groups 2/3 unused
#if defined(__clang_major__) && (__clang_major__ >= 23)
    tdm_i8 gz8 = {0, 0, 0, 0, 0, 0, 0, 0};
    __builtin_amdgcn_tensor_load_to_lds(g0, g1, gz, gz, gz8, 0);
#else
    __builtin_amdgcn_tensor_load_to_lds(g0, g1, gz, gz, 0);
#endif
    __builtin_amdgcn_s_wait_tensorcnt(0);               // s_wait_tensorcnt 0x0
  }
#else
  for (int i = threadIdx.x; i < DIM * DIM / 8; i += blockDim.x)
    ((v8h*)Bs)[i] = ((const v8h*)Bh)[i];
#endif
  __syncthreads();

  int wave = (blockIdx.x * blockDim.x + threadIdx.x) >> 5;
  int lane = threadIdx.x & 31;
  int m0 = wave * 2;
  if (m0 >= n_mtiles) return;         // wave-uniform: EXEC stays all-1s for WMMA
  bool two = (m0 + 1) < n_mtiles;
  int m1 = two ? (m0 + 1) : m0;

  int hi = lane >> 4, mr = lane & 15;
  const _Float16* arow0 = Xh + ((size_t)m0 * 16 + mr) * DIM;
  const _Float16* arow1 = Xh + ((size_t)m1 * 16 + mr) * DIM;

  v8f acc0[8] = {};
  v8f acc1[8] = {};
#pragma unroll
  for (int kt = 0; kt < 4; ++kt) {
    int kb = kt * 32 + hi * 8;
    // A frag (16-bit 16x32, ISA 7.12.2): halves 0..7 = K[kb..kb+7], 8..15 = K[kb+16..kb+23]
    v8h a0lo = *(const v8h*)(arow0 + kb);
    v8h a0hi = *(const v8h*)(arow0 + kb + 16);
    v8h a1lo = *(const v8h*)(arow1 + kb);
    v8h a1hi = *(const v8h*)(arow1 + kb + 16);
    v16h A0 = __builtin_shufflevector(a0lo, a0hi, 0, 1, 2, 3, 4, 5, 6, 7,
                                                  8, 9, 10, 11, 12, 13, 14, 15);
    v16h A1 = __builtin_shufflevector(a1lo, a1hi, 0, 1, 2, 3, 4, 5, 6, 7,
                                                  8, 9, 10, 11, 12, 13, 14, 15);
    const _Float16* brow = Bs + (size_t)kb * DIM + mr;  // B frag: lane -> N col, elems -> K (same order as A)
#pragma unroll
    for (int nt = 0; nt < 8; ++nt) {
      v16h B;
#pragma unroll
      for (int s = 0; s < 16; ++s)
        B[s] = brow[(size_t)((s & 7) + (s >> 3) * 16) * DIM + nt * 16];
      acc0[nt] = __builtin_amdgcn_wmma_f32_16x16x32_f16(
          false, A0, false, B, (short)0, acc0[nt], false, false);
      acc1[nt] = __builtin_amdgcn_wmma_f32_16x16x32_f16(
          false, A1, false, B, (short)0, acc1[nt], false, false);
    }
  }
  // C layout: VGPR i, lanes0-15 -> M=i, lanes16-31 -> M=i+8; N = lane&15
#pragma unroll
  for (int nt = 0; nt < 8; ++nt) {
    int col = nt * 16 + mr;
    float bias = vb[col];
#pragma unroll
    for (int i = 0; i < 8; ++i) {
      size_t row = (size_t)m0 * 16 + i + 8 * hi;
      V[row * DIM + col] = acc0[nt][i] + bias;
    }
    if (two) {
#pragma unroll
      for (int i = 0; i < 8; ++i) {
        size_t row = (size_t)m1 * 16 + i + 8 * hi;
        V[row * DIM + col] = acc1[nt][i] + bias;
      }
    }
  }
}

// segment max of score[src] over dst
__global__ void edge_max_kernel(const int* __restrict__ src, const int* __restrict__ dst,
                                const float* __restrict__ score, unsigned* mmax, int E) {
  int e = blockIdx.x * blockDim.x + threadIdx.x;
  if (e < E) uatomic_max(mmax + dst[e], ordf(score[src[e]]));
}

// w = exp(score[src] - m[dst]); ealpha[e] = w; ssum[dst] += w
__global__ void edge_exp_kernel(const int* __restrict__ src, const int* __restrict__ dst,
                                const float* __restrict__ score, const unsigned* __restrict__ mmax,
                                float* __restrict__ ealpha, float* ssum, int E) {
  int e = blockIdx.x * blockDim.x + threadIdx.x;
  if (e >= E) return;
  int d = dst[e];
  float w = __expf(score[src[e]] - unordf(mmax[d]));
  ealpha[e] = w;
  fatomic_add(ssum + d, w);
}

// O[dst] += (ealpha[e]/ssum[dst]) * V[src]   (one wave32 per edge; float4/lane)
__global__ void scatter_kernel(const int* __restrict__ src, const int* __restrict__ dst,
                               const float* __restrict__ ealpha, const float* __restrict__ ssum,
                               const float* __restrict__ V, float* O, int E) {
  int e = (blockIdx.x * blockDim.x + threadIdx.x) >> 5;
  int lane = threadIdx.x & 31;
  if (e >= E) return;
  int s = src[e], d = dst[e];
  float alpha = ealpha[e] / ssum[d];
  float4 v4 = ((const float4*)(V + (size_t)s * DIM))[lane];
  float* o = O + (size_t)d * DIM + lane * 4;
  fatomic_add(o + 0, alpha * v4.x);
  fatomic_add(o + 1, alpha * v4.y);
  fatomic_add(o + 2, alpha * v4.z);
  fatomic_add(o + 3, alpha * v4.w);
}

// ---------- host side ----------
static inline int cdiv(int a, int b) { return (a + b - 1) / b; }

static void run_layer(const float* X, const int* src, const int* dst,
                      const float* qw, const float* qb,
                      const float* kw, const float* kb,
                      const float* vw, const float* vb, const float* W,
                      float* O,
                      const float* deg, float* kred, float* u, float* c, float* score,
                      unsigned* mmax, float* ssum, float* ealpha,
                      _Float16* Xh, _Float16* Bh, float* V,
                      int N, int E, hipStream_t stream) {
  const float scale = 0.08838834764831845f;  // 1/sqrt(128)
  // node-side rank-1 score path
  fill_u32<<<1, 128, 0, stream>>>((unsigned*)kred, 0u, DIM);
  kred_kernel<<<cdiv(N, 64), 128, 0, stream>>>(X, deg, kred, N);
  vecmath_kernel<<<1, 128, 0, stream>>>(kred, kw, kb, W, qw, qb, (float)E, scale, u, c);
  score_kernel<<<cdiv(N, 8), 256, 0, stream>>>(X, u, c, score, N);
  // V projection via WMMA (f16 in, f32 accumulate), B staged to LDS by TDM
  pack_x_kernel<<<cdiv(N * DIM, 256), 256, 0, stream>>>(X, Xh, N * DIM);
  pack_w_kernel<<<cdiv(DIM * DIM, 256), 256, 0, stream>>>(vw, Bh);
  {
    int n_mtiles = N / 16;
    int waves = cdiv(n_mtiles, 2);
    vproj_wmma<<<cdiv(waves, 8), 256, 0, stream>>>(Xh, Bh, vb, V, n_mtiles);
  }
  // edge softmax + aggregation
  fill_u32<<<cdiv(N, 256), 256, 0, stream>>>(mmax, 0u, N);          // 0u == -inf (ordered map)
  fill_u32<<<cdiv(N, 256), 256, 0, stream>>>((unsigned*)ssum, 0u, N);
  edge_max_kernel<<<cdiv(E, 256), 256, 0, stream>>>(src, dst, score, mmax, E);
  edge_exp_kernel<<<cdiv(E, 256), 256, 0, stream>>>(src, dst, score, mmax, ealpha, ssum, E);
  fill_u32<<<cdiv(N * DIM, 256), 256, 0, stream>>>((unsigned*)O, 0u, N * DIM);
  scatter_kernel<<<cdiv(E, 8), 256, 0, stream>>>(src, dst, ealpha, ssum, V, O, E);
}

extern "C" void kernel_launch(void* const* d_in, const int* in_sizes, int n_in,
                              void* d_out, int out_size, void* d_ws, size_t ws_size,
                              hipStream_t stream) {
  const float* x  = (const float*)d_in[0];
  const int*   ei = (const int*)d_in[1];
  const float* q1w = (const float*)d_in[2];
  const float* k1w = (const float*)d_in[3];
  const float* v1w = (const float*)d_in[4];
  const float* W1  = (const float*)d_in[5];
  const float* q2w = (const float*)d_in[6];
  const float* k2w = (const float*)d_in[7];
  const float* v2w = (const float*)d_in[8];
  const float* W2  = (const float*)d_in[9];
  const float* q1b = (const float*)d_in[10];
  const float* k1b = (const float*)d_in[11];
  const float* v1b = (const float*)d_in[12];
  const float* q2b = (const float*)d_in[13];
  const float* k2b = (const float*)d_in[14];
  const float* v2b = (const float*)d_in[15];

  const int N = in_sizes[0] / DIM;      // 40000
  const int E = in_sizes[1] / 2;        // 640000
  const int* src = ei;
  const int* dst = ei + E;

  // workspace carve-out (256B aligned), ~55 MB total
  char* base = (char*)d_ws;
  size_t off = 0;
  auto alloc = [&](size_t bytes) -> char* {
    off = (off + 255) & ~(size_t)255;
    char* p = base + off;
    off += bytes;
    return p;
  };
  float*    deg    = (float*)alloc((size_t)N * 4);
  float*    kred   = (float*)alloc(DIM * 4);
  float*    u      = (float*)alloc(DIM * 4);
  float*    c      = (float*)alloc(4);
  float*    score  = (float*)alloc((size_t)N * 4);
  unsigned* mmax   = (unsigned*)alloc((size_t)N * 4);
  float*    ssum   = (float*)alloc((size_t)N * 4);
  float*    ealpha = (float*)alloc((size_t)E * 4);
  _Float16* Xh     = (_Float16*)alloc((size_t)N * DIM * 2);
  _Float16* Bh     = (_Float16*)alloc((size_t)DIM * DIM * 2);
  float*    V      = (float*)alloc((size_t)N * DIM * 4);
  float*    h      = (float*)alloc((size_t)N * DIM * 4);
  float*    out    = (float*)d_out;

  // degree counts (same for both layers)
  fill_u32<<<cdiv(N, 256), 256, 0, stream>>>((unsigned*)deg, 0u, N);
  deg_kernel<<<cdiv(E, 256), 256, 0, stream>>>(dst, deg, E);

  run_layer(x, src, dst, q1w, q1b, k1w, k1b, v1w, v1b, W1, h,
            deg, kred, u, c, score, mmax, ssum, ealpha, Xh, Bh, V, N, E, stream);
  run_layer(h, src, dst, q2w, q2b, k2w, k2b, v2w, v2b, W2, out,
            deg, kred, u, c, score, mmax, ssum, ealpha, Xh, Bh, V, N, E, stream);
}
